// Protein2LigandLayer_5781025980984
// MI455X (gfx1250) — matrix operand order repaired and checked by reading, not compiled, not run on hardware
//
#include <hip/hip_runtime.h>
#include <hip/hip_bf16.h>
#include <math.h>

#define H_    128
#define B_    4
#define L_    2048
#define K_    24
#define M_    64
#define FF_   512
#define NROW_ (B_*L_*K_)     // 196608
#define NTILE_ (NROW_/16)    // 12288
#define EPS_  1e-5f

typedef __attribute__((ext_vector_type(16))) _Float16 v16h;
typedef __attribute__((ext_vector_type(8)))  _Float16 v8h;
typedef __attribute__((ext_vector_type(8)))  float    v8f;

// ---------------------------------------------------------------- helpers ---

__device__ __forceinline__ v8f wmma16(v16h a, v16h b, v8f c) {
  // D = A(16x32 f16) x B(32x16 f16) + C(16x16 f32)
  return __builtin_amdgcn_wmma_f32_16x16x32_f16(false, a, false, b, (short)0, c,
                                                false, false);
}

__device__ __forceinline__ float gelu_f(float x) {
  return 0.5f * x * (1.0f + erff(x * 0.70710678118654752f));
}

// A-fragment built directly from the fp32 "concat" sources.
// kc in [0,12): cols [kc*32, kc*32+32) of [hY | hE | hV_bcast].
// lane<16 -> K {0..7,16..23} of the chunk; lane>=16 -> K {8..15,24..31}.
__device__ __forceinline__ v16h a_concat(const float* p0, const float* p1,
                                         const float* p2, int kc, int hs) {
  const float* src = (kc < 4) ? p0 : ((kc < 8) ? p1 : p2);
  int lc = (kc & 3) * 32 + hs * 8;
  const float4* q0 = (const float4*)(src + lc);
  const float4* q1 = (const float4*)(src + lc + 16);
  float4 u0 = q0[0], u1 = q0[1], u2 = q1[0], u3 = q1[1];
  v16h a;
  a[0]=(_Float16)u0.x; a[1]=(_Float16)u0.y; a[2]=(_Float16)u0.z; a[3]=(_Float16)u0.w;
  a[4]=(_Float16)u1.x; a[5]=(_Float16)u1.y; a[6]=(_Float16)u1.z; a[7]=(_Float16)u1.w;
  a[8]=(_Float16)u2.x; a[9]=(_Float16)u2.y; a[10]=(_Float16)u2.z; a[11]=(_Float16)u2.w;
  a[12]=(_Float16)u3.x; a[13]=(_Float16)u3.y; a[14]=(_Float16)u3.z; a[15]=(_Float16)u3.w;
  return a;
}

// A-fragment from an f16 LDS tile (stride in halves, colbase = kc*32 + hs*8).
__device__ __forceinline__ v16h a_lds(const _Float16* t, int stride, int m,
                                      int colbase) {
  const _Float16* p = t + m * stride + colbase;
  v8h lo = *(const v8h*)p;        // ds_load_b128
  v8h hi = *(const v8h*)(p + 16); // ds_load_b128
  v16h a;
  #pragma unroll
  for (int j = 0; j < 8; ++j) { a[j] = lo[j]; a[8 + j] = hi[j]; }
  return a;
}

// B-fragment from the pre-packed weight buffer (see prep_w layout).
__device__ __forceinline__ v16h b_frag(const _Float16* wf, int nt, int KC,
                                       int kc, int lane) {
  return *(const v16h*)(wf + ((size_t)(nt * KC + kc) * 32 + lane) * 16);
}

// D-frag (8 ntiles) -> gelu -> f16 LDS tile (cross-lane transpose).
__device__ __forceinline__ void store_tile_gelu(_Float16* tb, int stride,
                                                const v8f* acc, int m, int hs) {
  #pragma unroll
  for (int nt = 0; nt < 8; ++nt)
    #pragma unroll
    for (int v = 0; v < 8; ++v)
      tb[(v + 8 * hs) * stride + nt * 16 + m] = (_Float16)gelu_f(acc[nt][v]);
}

// D-frag (8 ntiles) -> raw f32 LDS tile [16][128].
__device__ __forceinline__ void store_tile_f32(float* mb, const v8f* acc,
                                               int m, int hs) {
  #pragma unroll
  for (int nt = 0; nt < 8; ++nt)
    #pragma unroll
    for (int v = 0; v < 8; ++v)
      mb[(v + 8 * hs) * H_ + nt * 16 + m] = acc[nt][v];
}

// LayerNorm row stats: lanes 0..15 each own one row of a [16][128] f32 tile.
__device__ __forceinline__ void row_stats(const float* buf, int lane,
                                          float* mu_s, float* rs_s) {
  if (lane < 16) {
    float s = 0.f, ss = 0.f;
    for (int c = 0; c < H_; ++c) { float v = buf[lane * H_ + c]; s += v; ss += v * v; }
    float mu = s * (1.0f / H_);
    float var = ss * (1.0f / H_) - mu * mu;
    mu_s[lane] = mu;
    rs_s[lane] = rsqrtf(var + EPS_);
  }
}

// -------------------------------------------------------------- weight prep -

// Repack W[Cout][Cin] (fp32 row-major) into f16 B-fragments:
// out[(((nt*KC + kc)*32 + lane)*16 + j)] = W[nt*16 + (lane&15)][kc*32 + (j>>3)*16 + (lane>>4)*8 + (j&7)]
__global__ __launch_bounds__(256) void prep_w(const float* __restrict__ W,
                                              _Float16* __restrict__ out,
                                              int Cout, int Cin) {
  int idx = blockIdx.x * blockDim.x + threadIdx.x;
  int total = Cout * Cin;
  if (idx >= total) return;
  int KC   = Cin >> 5;
  int j    = idx & 15;
  int lane = (idx >> 4) & 31;
  int fk   = idx >> 9;
  int kc   = fk % KC;
  int nt   = fk / KC;
  int n = nt * 16 + (lane & 15);
  int k = kc * 32 + (j >> 3) * 16 + (lane >> 4) * 8 + (j & 7);
  out[idx] = (_Float16)W[(size_t)n * Cin + k];
}

__global__ __launch_bounds__(256) void zero_f32(float* __restrict__ p, int n) {
  int i = blockIdx.x * blockDim.x + threadIdx.x;
  if (i < n) p[i] = 0.0f;
}

__global__ __launch_bounds__(256) void k_scale_dh(float* __restrict__ dh,
                                                  const float* __restrict__ ysc,
                                                  int n) {
  int i = blockIdx.x * blockDim.x + threadIdx.x;
  if (i < n) dh[i] = dh[i] / ysc[i >> 7];  // i/H -> b*M + m
}

// ---------------------------------------------- kernel: MLP1 + atomic scatter

__global__ __launch_bounds__(128) void k_mlp_scatter(
    const float* __restrict__ hY, const float* __restrict__ hE,
    const float* __restrict__ hV, const int* __restrict__ nn,
    const _Float16* __restrict__ wf1, const _Float16* __restrict__ wf2,
    const _Float16* __restrict__ wf3,
    const float* __restrict__ b1, const float* __restrict__ b2,
    const float* __restrict__ b3, float* __restrict__ dh) {
  __shared__ _Float16 tb_s[4][16 * 136];
  __shared__ float    mb_s[4][16 * H_];
  __shared__ int      nn_s[4][16];

  int wave = threadIdx.x >> 5, lane = threadIdx.x & 31;
  int tile = blockIdx.x * 4 + wave;
  int gbase = tile * 16;
  int m = lane & 15, hs = lane >> 4;
  size_t grow = (size_t)(gbase + m);
  const float* p0 = hY + grow * H_;
  const float* p1 = hE + grow * H_;
  const float* p2 = hV + (grow / K_) * H_;
  _Float16* tb = tb_s[wave];
  float*    mb = mb_s[wave];

  if (lane < 16) nn_s[wave][lane] = nn[gbase + lane];

  v8f acc[8];
  // ---- layer 1: 3H -> H, gelu -------------------------------------------
  #pragma unroll
  for (int nt = 0; nt < 8; ++nt) {
    float bv = b1[nt * 16 + m];
    v8f c; 
    #pragma unroll
    for (int v = 0; v < 8; ++v) c[v] = bv;
    #pragma unroll
    for (int kc = 0; kc < 12; ++kc)
      c = wmma16(a_concat(p0, p1, p2, kc, hs), b_frag(wf1, nt, 12, kc, lane), c);
    acc[nt] = c;
  }
  store_tile_gelu(tb, 136, acc, m, hs);
  __syncthreads();
  // ---- layer 2: H -> H, gelu --------------------------------------------
  #pragma unroll
  for (int nt = 0; nt < 8; ++nt) {
    float bv = b2[nt * 16 + m];
    v8f c;
    #pragma unroll
    for (int v = 0; v < 8; ++v) c[v] = bv;
    #pragma unroll
    for (int kc = 0; kc < 4; ++kc)
      c = wmma16(a_lds(tb, 136, m, kc * 32 + hs * 8), b_frag(wf2, nt, 4, kc, lane), c);
    acc[nt] = c;
  }
  __syncthreads();
  store_tile_gelu(tb, 136, acc, m, hs);
  __syncthreads();
  // ---- layer 3: H -> H (no gelu) ----------------------------------------
  #pragma unroll
  for (int nt = 0; nt < 8; ++nt) {
    float bv = b3[nt * 16 + m];
    v8f c;
    #pragma unroll
    for (int v = 0; v < 8; ++v) c[v] = bv;
    #pragma unroll
    for (int kc = 0; kc < 4; ++kc)
      c = wmma16(a_lds(tb, 136, m, kc * 32 + hs * 8), b_frag(wf3, nt, 4, kc, lane), c);
    acc[nt] = c;
  }
  store_tile_f32(mb, acc, m, hs);
  __syncthreads();
  // ---- scatter-add into dh[b, nn_idx, :] --------------------------------
  int bI = gbase / (L_ * K_);
  for (int e = lane; e < 16 * H_; e += 32) {
    int r = e >> 7, cl = e & 127;
    atomicAdd(dh + ((size_t)(bI * M_ + nn_s[wave][r]) * H_ + cl), mb[e]);
  }
}

// ------------------------------------- kernel: gather + LN1 + FFN + LN2 -> hY

__global__ __launch_bounds__(64) void k_part2a(
    const float* __restrict__ hYn, const int* __restrict__ nn,
    const float* __restrict__ dh,
    const _Float16* __restrict__ wfin, const _Float16* __restrict__ wfout,
    const float* __restrict__ bin, const float* __restrict__ bout,
    const float* __restrict__ n1s, const float* __restrict__ n1b,
    const float* __restrict__ n2s, const float* __restrict__ n2b,
    float* __restrict__ outY) {
  __shared__ float    hy_s[2][16 * H_];
  __shared__ _Float16 hyh_s[2][16 * 136];
  __shared__ _Float16 tbh_s[2][16 * 264];
  __shared__ float    mb_s[2][16 * H_];
  __shared__ float    mu_s[2][16], rs_s[2][16];

  int wave = threadIdx.x >> 5, lane = threadIdx.x & 31;
  int tile = blockIdx.x * 2 + wave;
  int gbase = tile * 16;
  int m = lane & 15, hs = lane >> 4;
  float*    hy  = hy_s[wave];
  _Float16* hyh = hyh_s[wave];
  _Float16* tbh = tbh_s[wave];
  float*    mb  = mb_s[wave];

  // ---- gather dh + residual ---------------------------------------------
  int bI = gbase / (L_ * K_);
  for (int e = lane; e < 16 * H_; e += 32) {
    int r = e >> 7, cl = e & 127;
    size_t g = (size_t)(gbase + r);
    int mm = nn[g];
    hy[e] = hYn[g * H_ + cl] + dh[((size_t)(bI * M_ + mm)) * H_ + cl];
  }
  __syncthreads();
  // ---- LN1 ---------------------------------------------------------------
  row_stats(hy, lane, mu_s[wave], rs_s[wave]);
  __syncthreads();
  for (int e = lane; e < 16 * H_; e += 32) {
    int r = e >> 7, cl = e & 127;
    float v = (hy[e] - mu_s[wave][r]) * rs_s[wave][r] * n1s[cl] + n1b[cl];
    hy[e] = v;
    hyh[r * 136 + cl] = (_Float16)v;
  }
  __syncthreads();
  // ---- FFN: Win (H->FF, gelu), Wout (FF->H), FF in 2 halves of 256 -------
  v8f accO[8];
  #pragma unroll
  for (int nt = 0; nt < 8; ++nt) {
    float bv = bout[nt * 16 + m];
    #pragma unroll
    for (int v = 0; v < 8; ++v) accO[nt][v] = bv;
  }
  for (int half = 0; half < 2; ++half) {
    for (int nt2 = 0; nt2 < 16; ++nt2) {
      int gnt = half * 16 + nt2;
      float bv = bin[gnt * 16 + m];
      v8f c;
      #pragma unroll
      for (int v = 0; v < 8; ++v) c[v] = bv;
      #pragma unroll
      for (int kc = 0; kc < 4; ++kc)
        c = wmma16(a_lds(hyh, 136, m, kc * 32 + hs * 8),
                   b_frag(wfin, gnt, 4, kc, lane), c);
      #pragma unroll
      for (int v = 0; v < 8; ++v)
        tbh[(v + 8 * hs) * 264 + nt2 * 16 + m] = (_Float16)gelu_f(c[v]);
    }
    __syncthreads();
    #pragma unroll
    for (int nt = 0; nt < 8; ++nt)
      #pragma unroll
      for (int kc = 0; kc < 8; ++kc)
        accO[nt] = wmma16(a_lds(tbh, 264, m, kc * 32 + hs * 8),
                          b_frag(wfout, nt, 16, half * 8 + kc, lane), accO[nt]);
    __syncthreads();
  }
  store_tile_f32(mb, accO, m, hs);
  __syncthreads();
  // ---- residual + LN2 -> d_out (h_Y) ------------------------------------
  for (int e = lane; e < 16 * H_; e += 32) mb[e] += hy[e];
  __syncthreads();
  row_stats(mb, lane, mu_s[wave], rs_s[wave]);
  __syncthreads();
  for (int e = lane; e < 16 * H_; e += 32) {
    int r = e >> 7, cl = e & 127;
    outY[(size_t)(gbase + r) * H_ + cl] =
        (mb[e] - mu_s[wave][r]) * rs_s[wave][r] * n2s[cl] + n2b[cl];
  }
}

// ------------------------------------------ kernel: MLP2 + LN3 -> h_E output

__global__ __launch_bounds__(128) void k_mlp2_ln(
    const float* __restrict__ hYf, const float* __restrict__ hE,
    const float* __restrict__ hV,
    const _Float16* __restrict__ wf1, const _Float16* __restrict__ wf2,
    const _Float16* __restrict__ wf3,
    const float* __restrict__ b1, const float* __restrict__ b2,
    const float* __restrict__ b3,
    const float* __restrict__ ls, const float* __restrict__ lb,
    float* __restrict__ outE) {
  __shared__ _Float16 tb_s[4][16 * 136];
  __shared__ float    mb_s[4][16 * H_];
  __shared__ float    mu_s[4][16], rs_s[4][16];

  int wave = threadIdx.x >> 5, lane = threadIdx.x & 31;
  int tile = blockIdx.x * 4 + wave;
  int gbase = tile * 16;
  int m = lane & 15, hs = lane >> 4;
  size_t grow = (size_t)(gbase + m);
  const float* p0 = hYf + grow * H_;
  const float* p1 = hE + grow * H_;
  const float* p2 = hV + (grow / K_) * H_;
  _Float16* tb = tb_s[wave];
  float*    mb = mb_s[wave];

  v8f acc[8];
  #pragma unroll
  for (int nt = 0; nt < 8; ++nt) {
    float bv = b1[nt * 16 + m];
    v8f c;
    #pragma unroll
    for (int v = 0; v < 8; ++v) c[v] = bv;
    #pragma unroll
    for (int kc = 0; kc < 12; ++kc)
      c = wmma16(a_concat(p0, p1, p2, kc, hs), b_frag(wf1, nt, 12, kc, lane), c);
    acc[nt] = c;
  }
  store_tile_gelu(tb, 136, acc, m, hs);
  __syncthreads();
  #pragma unroll
  for (int nt = 0; nt < 8; ++nt) {
    float bv = b2[nt * 16 + m];
    v8f c;
    #pragma unroll
    for (int v = 0; v < 8; ++v) c[v] = bv;
    #pragma unroll
    for (int kc = 0; kc < 4; ++kc)
      c = wmma16(a_lds(tb, 136, m, kc * 32 + hs * 8), b_frag(wf2, nt, 4, kc, lane), c);
    acc[nt] = c;
  }
  __syncthreads();
  store_tile_gelu(tb, 136, acc, m, hs);
  __syncthreads();
  #pragma unroll
  for (int nt = 0; nt < 8; ++nt) {
    float bv = b3[nt * 16 + m];
    v8f c;
    #pragma unroll
    for (int v = 0; v < 8; ++v) c[v] = bv;
    #pragma unroll
    for (int kc = 0; kc < 4; ++kc)
      c = wmma16(a_lds(tb, 136, m, kc * 32 + hs * 8), b_frag(wf3, nt, 4, kc, lane), c);
    acc[nt] = c;
  }
  store_tile_f32(mb, acc, m, hs);
  __syncthreads();
  // residual + LN3
  for (int e = lane; e < 16 * H_; e += 32) {
    int r = e >> 7, cl = e & 127;
    mb[e] += hE[(size_t)(gbase + r) * H_ + cl];
  }
  __syncthreads();
  row_stats(mb, lane, mu_s[wave], rs_s[wave]);
  __syncthreads();
  for (int e = lane; e < 16 * H_; e += 32) {
    int r = e >> 7, cl = e & 127;
    outE[(size_t)(gbase + r) * H_ + cl] =
        (mb[e] - mu_s[wave][r]) * rs_s[wave][r] * ls[cl] + lb[cl];
  }
}

// ------------------------------------------------------------------ launcher

extern "C" void kernel_launch(void* const* d_in, const int* in_sizes, int n_in,
                              void* d_out, int out_size, void* d_ws,
                              size_t ws_size, hipStream_t stream) {
  (void)in_sizes; (void)n_in; (void)out_size; (void)ws_size;

  const int*   nn   = (const int*)d_in[0];
  const float* Ysc  = (const float*)d_in[1];
  const float* hY   = (const float*)d_in[2];
  const float* hE   = (const float*)d_in[3];
  const float* hV   = (const float*)d_in[4];
  const float* W1w  = (const float*)d_in[5],  *W1b  = (const float*)d_in[6];
  const float* W2w  = (const float*)d_in[7],  *W2b  = (const float*)d_in[8];
  const float* W3w  = (const float*)d_in[9],  *W3b  = (const float*)d_in[10];
  const float* W11w = (const float*)d_in[11], *W11b = (const float*)d_in[12];
  const float* W12w = (const float*)d_in[13], *W12b = (const float*)d_in[14];
  const float* W13w = (const float*)d_in[15], *W13b = (const float*)d_in[16];
  const float* Winw = (const float*)d_in[17], *Winb = (const float*)d_in[18];
  const float* Woutw= (const float*)d_in[19], *Woutb= (const float*)d_in[20];
  const float* n1s  = (const float*)d_in[21], *n1b  = (const float*)d_in[22];
  const float* n2s  = (const float*)d_in[23], *n2b  = (const float*)d_in[24];
  const float* n3s  = (const float*)d_in[25], *n3b  = (const float*)d_in[26];

  char* ws = (char*)d_ws;
  float*    dh   = (float*)(ws + 0);                 // B*M*H f32 = 131072 B
  _Float16* wf1  = (_Float16*)(ws + 131072);         // 128x384 -> 98304 B
  _Float16* wf2  = (_Float16*)(ws + 229376);         // 128x128 -> 32768 B
  _Float16* wf3  = (_Float16*)(ws + 262144);         // 128x128 -> 32768 B
  _Float16* wf11 = (_Float16*)(ws + 294912);         // 128x384 -> 98304 B
  _Float16* wf12 = (_Float16*)(ws + 393216);         // 128x128 -> 32768 B
  _Float16* wf13 = (_Float16*)(ws + 425984);         // 128x128 -> 32768 B
  _Float16* wfin = (_Float16*)(ws + 458752);         // 512x128 -> 131072 B
  _Float16* wfout= (_Float16*)(ws + 589824);         // 128x512 -> 131072 B

  const int T = 256;
  auto gp = [](int n) { return (n + 255) / 256; };

  prep_w<<<gp(128 * 384), T, 0, stream>>>(W1w,  wf1,  128, 384);
  prep_w<<<gp(128 * 128), T, 0, stream>>>(W2w,  wf2,  128, 128);
  prep_w<<<gp(128 * 128), T, 0, stream>>>(W3w,  wf3,  128, 128);
  prep_w<<<gp(128 * 384), T, 0, stream>>>(W11w, wf11, 128, 384);
  prep_w<<<gp(128 * 128), T, 0, stream>>>(W12w, wf12, 128, 128);
  prep_w<<<gp(128 * 128), T, 0, stream>>>(W13w, wf13, 128, 128);
  prep_w<<<gp(512 * 128), T, 0, stream>>>(Winw, wfin, 512, 128);
  prep_w<<<gp(128 * 512), T, 0, stream>>>(Woutw,wfout,128, 512);

  zero_f32<<<gp(B_ * M_ * H_), T, 0, stream>>>(dh, B_ * M_ * H_);

  k_mlp_scatter<<<NTILE_ / 4, 128, 0, stream>>>(hY, hE, hV, nn, wf1, wf2, wf3,
                                                W1b, W2b, W3b, dh);

  k_scale_dh<<<gp(B_ * M_ * H_), T, 0, stream>>>(dh, Ysc, B_ * M_ * H_);

  float* outY = (float*)d_out;
  float* outE = outY + (size_t)NROW_ * H_;

  k_part2a<<<NTILE_ / 2, 64, 0, stream>>>(hY, nn, dh, wfin, wfout, Winb, Woutb,
                                          n1s, n1b, n2s, n2b, outY);

  k_mlp2_ln<<<NTILE_ / 4, 128, 0, stream>>>(outY, hE, hV, wf11, wf12, wf13,
                                            W11b, W12b, W13b, n3s, n3b, outE);
}
